// RoCrossViViT_bis_54443005444620
// MI455X (gfx1250) — compile-verified, matrix operand order checked by reading, not run on hardware
//
#include <hip/hip_runtime.h>

// ---------------------------------------------------------------------------
// RoCrossViViT forward for MI455X (gfx1250, wave32, WMMA).
// Dense math -> one batched WMMA GEMM (v_wmma_f32_16x16x32_f16, f32 acc).
// LDS tiles: A row-major [m][k], B transposed [n][k], row stride 40 halves
// (80B, 16B aligned) so every lane's fragment = two ds_load_b128.
// Global staging: float4 (b128) fast path, branchless clamped edge path.
// ---------------------------------------------------------------------------

typedef __attribute__((ext_vector_type(16))) _Float16 v16h;
typedef __attribute__((ext_vector_type(8)))  _Float16 v8h;
typedef __attribute__((ext_vector_type(8)))  float    v8f;
typedef __attribute__((ext_vector_type(4)))  float    v4f;

__constant__ float c_freqs[4] = {12.0f, 31.0f, 24.0f, 60.0f};

#define TM 64
#define TN 64
#define TK 32
#define LDH 40  // halves per LDS row: 80 bytes, multiple of 16B

// ---------------------------------------------------------------------------
// Batched WMMA GEMM: C[z] = act( A[z] x B[z] + bias + Res )
// Batch offset for X in {A,B,C}: (z / hX) * sXo + (z % hX) * sXh
// transB==0: B is [K,N]; transB==1: B is [N,K].
// ---------------------------------------------------------------------------
__global__ __launch_bounds__(256) void rv_gemm(
    const float* __restrict__ A, long ldA, long sAo, long sAh, int hA,
    const float* __restrict__ Bm, long ldB, long sBo, long sBh, int hB, int transB,
    float* __restrict__ C, long ldC, long sCo, long sCh, int hC,
    const float* __restrict__ bias, const float* __restrict__ Res, long ldR,
    int M, int N, int K, int act)
{
    __shared__ alignas(16) _Float16 As[TM][LDH];
    __shared__ alignas(16) _Float16 Bt[TN][LDH];   // transposed: [n][k]

    const int z = blockIdx.z;
    const float* Ab = A  + (long)(z / hA) * sAo + (long)(z % hA) * sAh;
    const float* Bb = Bm + (long)(z / hB) * sBo + (long)(z % hB) * sBh;
    float*       Cb = C  + (long)(z / hC) * sCo + (long)(z % hC) * sCh;

    const int m0 = blockIdx.x * TM;
    const int n0 = blockIdx.y * TN;
    const int tid  = threadIdx.x;
    const int lane = tid & 31;
    const int wave = tid >> 5;
    const int wm = wave & 3;    // 4 row tiles of 16
    const int wn = wave >> 2;   // 2 col groups of 32 (2 tiles each)
    const int l16 = lane & 15;
    const int hlf = lane >> 4;

    const bool fullM = (m0 + TM) <= M;
    const bool fullN = (n0 + TN) <= N;
    const bool vecA  = fullM && ((ldA & 3) == 0);
    const bool vecB  = fullN && ((ldB & 3) == 0);

    v8f acc0 = {};
    v8f acc1 = {};

    for (int k0 = 0; k0 < K; k0 += TK) {
        const bool fullK = (k0 + TK) <= K;
        // prefetch next K tile of A
        if (k0 + TK < K) {
            const float* pf = Ab + (long)(m0 + (tid >> 2)) * ldA + (k0 + TK);
            __builtin_prefetch(pf, 0, 1);
        }

        // ---- stage A 64x32 ----
        if (vecA && fullK) {
            const int row = tid >> 2, ks = (tid & 3) * 8;
            const float* src = Ab + (long)(m0 + row) * ldA + (k0 + ks);
            v4f f0 = *(const v4f*)src;
            v4f f1 = *(const v4f*)(src + 4);
            v8h h;
#pragma unroll
            for (int i = 0; i < 4; ++i) { h[i] = (_Float16)f0[i]; h[i + 4] = (_Float16)f1[i]; }
            *(v8h*)&As[row][ks] = h;
        } else {
            for (int i = tid; i < TM * TK; i += 256) {
                const int m = i >> 5, k = i & 31;
                const int gm = m0 + m, gk = k0 + k;
                const int cm = gm < M ? gm : (M - 1);
                const int ck = gk < K ? gk : (K - 1);
                float v = Ab[(long)cm * ldA + ck];            // always load, clamped
                v = (gm < M && gk < K) ? v : 0.0f;            // branchless mask
                As[m][k] = (_Float16)v;
            }
        }

        // ---- stage B 32x64 into Bt[n][k] ----
        if (transB) {
            if (vecB && fullK) {
                const int n = tid >> 2, ks = (tid & 3) * 8;
                const float* src = Bb + (long)(n0 + n) * ldB + (k0 + ks);
                v4f f0 = *(const v4f*)src;
                v4f f1 = *(const v4f*)(src + 4);
                v8h h;
#pragma unroll
                for (int i = 0; i < 4; ++i) { h[i] = (_Float16)f0[i]; h[i + 4] = (_Float16)f1[i]; }
                *(v8h*)&Bt[n][ks] = h;
            } else {
                for (int i = tid; i < TN * TK; i += 256) {
                    const int n = i >> 5, k = i & 31;
                    const int gn = n0 + n, gk = k0 + k;
                    const int cn = gn < N ? gn : (N - 1);
                    const int ck = gk < K ? gk : (K - 1);
                    float v = Bb[(long)cn * ldB + ck];
                    v = (gn < N && gk < K) ? v : 0.0f;
                    Bt[n][k] = (_Float16)v;
                }
            }
        } else {
            if (vecB && fullK) {
                const int k = tid >> 3, ns = (tid & 7) * 8;
                const float* src = Bb + (long)(k0 + k) * ldB + (n0 + ns);
                v4f f0 = *(const v4f*)src;
                v4f f1 = *(const v4f*)(src + 4);
#pragma unroll
                for (int i = 0; i < 4; ++i) {
                    Bt[ns + i][k]     = (_Float16)f0[i];
                    Bt[ns + 4 + i][k] = (_Float16)f1[i];
                }
            } else {
                for (int i = tid; i < TN * TK; i += 256) {
                    const int n = i >> 5, k = i & 31;
                    const int gn = n0 + n, gk = k0 + k;
                    const int cn = gn < N ? gn : (N - 1);
                    const int ck = gk < K ? gk : (K - 1);
                    float v = Bb[(long)ck * ldB + cn];
                    v = (gn < N && gk < K) ? v : 0.0f;
                    Bt[n][k] = (_Float16)v;
                }
            }
        }
        __syncthreads();

        // ---- fragments (ISA 7.12.2): two b128 LDS reads per fragment ----
        const int arow = wm * 16 + l16;
        v8h a0 = *(const v8h*)&As[arow][hlf * 8];
        v8h a1 = *(const v8h*)&As[arow][16 + hlf * 8];
        v16h af;
#pragma unroll
        for (int e = 0; e < 8; ++e) { af[e] = a0[e]; af[e + 8] = a1[e]; }

        const int br0 = wn * 32 + l16;
        v8h b0 = *(const v8h*)&Bt[br0][hlf * 8];
        v8h b1 = *(const v8h*)&Bt[br0][16 + hlf * 8];
        v16h bf0;
#pragma unroll
        for (int e = 0; e < 8; ++e) { bf0[e] = b0[e]; bf0[e + 8] = b1[e]; }

        const int br1 = br0 + 16;
        v8h b2 = *(const v8h*)&Bt[br1][hlf * 8];
        v8h b3 = *(const v8h*)&Bt[br1][16 + hlf * 8];
        v16h bf1;
#pragma unroll
        for (int e = 0; e < 8; ++e) { bf1[e] = b2[e]; bf1[e + 8] = b3[e]; }

        acc0 = __builtin_amdgcn_wmma_f32_16x16x32_f16(false, af, false, bf0, (short)0, acc0, false, false);
        acc1 = __builtin_amdgcn_wmma_f32_16x16x32_f16(false, af, false, bf1, (short)0, acc1, false, false);
        __syncthreads();
    }

    // ---- epilogue: C/D layout — VGPR r: M = r + (lane/16)*8, N = lane%16 ----
#pragma unroll
    for (int r = 0; r < 8; ++r) {
        const int m  = m0 + wm * 16 + r + hlf * 8;
        const int na = n0 + wn * 32 + l16;
        const int nb = na + 16;
        if (m < M && na < N) {
            float v = acc0[r];
            if (bias) v += bias[na];
            if (Res)  v += Res[(long)m * ldR + na];
            if (act == 1) v = fmaxf(v, 0.0f);
            Cb[(long)m * ldC + na] = v;
        }
        if (m < M && nb < N) {
            float v = acc1[r];
            if (bias) v += bias[nb];
            if (Res)  v += Res[(long)m * ldR + nb];
            if (act == 1) v = fmaxf(v, 0.0f);
            Cb[(long)m * ldC + nb] = v;
        }
    }
}

// ---------------------------------------------------------------------------
// Support kernels
// ---------------------------------------------------------------------------
__global__ __launch_bounds__(128) void rv_layernorm(
    const float* __restrict__ in, float* __restrict__ out,
    const float* __restrict__ g, const float* __restrict__ b, int dim)
{
    __shared__ float red[128];
    const long row = blockIdx.x;
    const float* x = in + row * (long)dim;
    const int tid = threadIdx.x;
    float s = 0.0f;
    for (int c = tid; c < dim; c += 128) s += x[c];
    red[tid] = s; __syncthreads();
    for (int k = 64; k > 0; k >>= 1) { if (tid < k) red[tid] += red[tid + k]; __syncthreads(); }
    const float mean = red[0] / dim; __syncthreads();
    float v = 0.0f;
    for (int c = tid; c < dim; c += 128) { float d = x[c] - mean; v += d * d; }
    red[tid] = v; __syncthreads();
    for (int k = 64; k > 0; k >>= 1) { if (tid < k) red[tid] += red[tid + k]; __syncthreads(); }
    const float inv = rsqrtf(red[0] / dim + 1e-5f);
    for (int c = tid; c < dim; c += 128)
        out[row * (long)dim + c] = (x[c] - mean) * inv * g[c] + b[c];
}

__global__ __launch_bounds__(256) void rv_softmax(float* __restrict__ buf, int ncols, float scale)
{
    __shared__ float red[256];
    const long row = blockIdx.x;
    float* x = buf + row * (long)ncols;
    const int tid = threadIdx.x;
    float v = (tid < ncols) ? x[tid] * scale : -3.4e38f;
    red[tid] = v; __syncthreads();
    for (int k = 128; k > 0; k >>= 1) { if (tid < k) red[tid] = fmaxf(red[tid], red[tid + k]); __syncthreads(); }
    const float m = red[0]; __syncthreads();
    float e = (tid < ncols) ? __expf(v - m) : 0.0f;
    red[tid] = e; __syncthreads();
    for (int k = 128; k > 0; k >>= 1) { if (tid < k) red[tid] += red[tid + k]; __syncthreads(); }
    if (tid < ncols) x[tid] = e / red[0];
}

// Patchified A matrix: rows = bt*256+n, cols f = (py*8+px)*11 + c
__global__ void rv_build_patchA(const float* __restrict__ ctx, const float* __restrict__ tc,
                                float* __restrict__ A, int row0, int nrows)
{
    long idx = blockIdx.x * (long)blockDim.x + threadIdx.x;
    if (idx >= (long)nrows * 704) return;
    int f = (int)(idx % 704);
    long lr = idx / 704;
    int row = row0 + (int)lr;
    int bt = row >> 8, n = row & 255;
    int c = f % 11, pp = f / 11;
    int py = pp >> 3, px = pp & 7;
    int g1 = n >> 4, g2 = n & 15;
    int r = g1 * 8 + py, ci = g2 * 8 + px;
    float v;
    if (c < 3) {
        v = ctx[((long)(bt * 3 + c)) * 16384 + r * 128 + ci];
    } else {
        int i = (c - 3) >> 1;
        float a = 6.2831853072f * tc[((long)(bt * 4 + i)) * 16384 + r * 128 + ci] / c_freqs[i];
        v = ((c - 3) & 1) ? __cosf(a) : __sinf(a);
    }
    A[lr * 704 + f] = v;
}

__global__ void rv_build_tsA(const float* __restrict__ ts, const float* __restrict__ tc,
                             float* __restrict__ A)
{
    long idx = blockIdx.x * (long)blockDim.x + threadIdx.x;
    if (idx >= 192L * 11) return;
    int c = (int)(idx % 11);
    int row = (int)(idx / 11);
    float v;
    if (c < 3) v = ts[row * 3 + c];
    else {
        int i = (c - 3) >> 1;
        float a = 6.2831853072f * tc[((long)(row * 4 + i)) * 16384] / c_freqs[i];
        v = ((c - 3) & 1) ? __cosf(a) : __sinf(a);
    }
    A[row * 11 + c] = v;
}

// Axial RoPE tables: sin/cos[(b*tok + t)*32 + d], tok = GP*GP
__global__ void rv_rope_tables(const float* __restrict__ coords, const float* __restrict__ scales,
                               float* __restrict__ sinT, float* __restrict__ cosT, int GP)
{
    int tok = GP * GP;
    long idx = blockIdx.x * (long)blockDim.x + threadIdx.x;
    long total = 4L * tok * 32;
    if (idx >= total) return;
    int d = (int)(idx & 31);
    long rem = idx >> 5;
    int t = (int)(rem % tok);
    int b = (int)(rem / tok);
    int wd = t / GP, hd = t % GP;
    float cv = (d < 16) ? coords[(long)b * 2 * tok + wd]
                        : coords[(((long)b * 2 + 1) * GP + hd) * (long)GP];
    float ang = cv * scales[d & 15] * 3.14159265359f;
    sinT[idx] = __sinf(ang);
    cosT[idx] = __cosf(ang);
}

// Apply RoPE to a 256-wide (4 heads x 64) section starting at col0.
__global__ void rv_rope_apply(float* __restrict__ buf, long rows, long rstride, int col0,
                              const float* __restrict__ sinT, const float* __restrict__ cosT,
                              int rows_per_b, int tok_per_b)
{
    long idx = blockIdx.x * (long)blockDim.x + threadIdx.x;
    if (idx >= rows * 128) return;
    int p = (int)(idx & 127);
    long row = idx >> 7;
    int head = p >> 5, jp = p & 31;
    int b = (int)(row / rows_per_b);
    int tok = (int)(row % tok_per_b);
    long ti = ((long)(b * tok_per_b + tok)) * 32 + jp;
    float s = sinT[ti], c = cosT[ti];
    float* q = buf + row * rstride + col0 + head * 64 + 2 * jp;
    float x1 = q[0], x2 = q[1];
    q[0] = x1 * c - x2 * s;
    q[1] = x2 * c + x1 * s;
}

__global__ void rv_add_pos(float* __restrict__ x, const float* __restrict__ pos,
                           int rows, int dim, int T)
{
    long idx = blockIdx.x * (long)blockDim.x + threadIdx.x;
    if (idx >= (long)rows * dim) return;
    int c = (int)(idx % dim);
    int row = (int)(idx / dim);
    x[idx] += pos[(row % T) * dim + c];
}

__device__ __forceinline__ float rv_gelu(float x)
{
    return 0.5f * x * (1.0f + erff(x * 0.70710678f));
}

__global__ void rv_glu(const float* __restrict__ u, float* __restrict__ t, long total, int half)
{
    long idx = blockIdx.x * (long)blockDim.x + threadIdx.x;
    if (idx >= total) return;
    long row = idx / half;
    int i = (int)(idx % half);
    float a = u[row * 2 * half + i];
    float g = u[row * 2 * half + half + i];
    t[row * half + i] = a * rv_gelu(g);
}

__global__ void rv_gelu_inplace(float* __restrict__ u, long n)
{
    long idx = blockIdx.x * (long)blockDim.x + threadIdx.x;
    if (idx < n) u[idx] = rv_gelu(u[idx]);
}

// One-query cross attention per (bt, h): q [192,256], kv [49152,512]
__global__ __launch_bounds__(256) void rv_cross_attn(
    const float* __restrict__ q, const float* __restrict__ kv,
    float* __restrict__ o, float* __restrict__ probs, int H)
{
    __shared__ float qs[64];
    __shared__ float lo[256];
    __shared__ float red[256];
    const int bh = blockIdx.x;
    const int bt = bh / H, h = bh % H;
    const int tid = threadIdx.x;
    if (tid < 64) qs[tid] = q[(long)bt * 256 + h * 64 + tid];
    __syncthreads();
    const float* krow = kv + ((long)bt * 256 + tid) * 512 + h * 64;
    float acc = 0.0f;
#pragma unroll 8
    for (int d = 0; d < 64; ++d) acc += qs[d] * krow[d];
    float v = acc * 0.125f;
    lo[tid] = v; red[tid] = v; __syncthreads();
    for (int k = 128; k > 0; k >>= 1) { if (tid < k) red[tid] = fmaxf(red[tid], red[tid + k]); __syncthreads(); }
    float m = red[0]; __syncthreads();
    float e = __expf(v - m);
    red[tid] = e; __syncthreads();
    for (int k = 128; k > 0; k >>= 1) { if (tid < k) red[tid] += red[tid + k]; __syncthreads(); }
    float p = e / red[0];
    lo[tid] = p;
    probs[(long)bh * 256 + tid] = p;
    __syncthreads();
    if (tid < 64) {
        float a = 0.0f;
        for (int j = 0; j < 256; ++j)
            a += lo[j] * kv[((long)bt * 256 + j) * 512 + 256 + h * 64 + tid];
        o[(long)bt * 256 + h * 64 + tid] = a;
    }
}

// conv1d (k=3, pad 1) over NLC layout, fused ReLU
__global__ void rv_conv1d_relu(const float* __restrict__ in, const float* __restrict__ Wt,
                               const float* __restrict__ bias, float* __restrict__ out,
                               int Bn, int L, int Cin, int Cout)
{
    long idx = blockIdx.x * (long)blockDim.x + threadIdx.x;
    long total = (long)Bn * L * Cout;
    if (idx >= total) return;
    int o = (int)(idx % Cout);
    long rem = idx / Cout;
    int t = (int)(rem % L);
    int b = (int)(rem / L);
    float acc = bias[o];
    for (int dt = 0; dt < 3; ++dt) {
        int tt = t + dt - 1;
        if (tt < 0 || tt >= L) continue;
        const float* ip = in + ((long)b * L + tt) * Cin;
        const float* wp = Wt + (long)o * Cin * 3 + dt;
        for (int i = 0; i < Cin; ++i) acc += ip[i] * wp[(long)i * 3];
    }
    out[((long)b * L + t) * Cout + o] = fmaxf(acc, 0.0f);
}

// ---------------------------------------------------------------------------
// Host-side helpers and orchestration
// ---------------------------------------------------------------------------
static inline void mm(hipStream_t st, const float* A, long ldA, const float* W, float* C,
                      int M, int N, int K, const float* bias, const float* Res, int act)
{
    dim3 g((M + TM - 1) / TM, (N + TN - 1) / TN, 1);
    rv_gemm<<<g, 256, 0, st>>>(A, ldA, 0, 0, 1, W, N, 0, 0, 1, 0,
                               C, N, 0, 0, 1, bias, Res, N, M, N, K, act);
}

static inline void gemmB(hipStream_t st,
                         const float* A, long ldA, long sAo, long sAh, int hA,
                         const float* Bm, long ldB, long sBo, long sBh, int hB, int transB,
                         float* C, long ldC, long sCo, long sCh, int hC,
                         int M, int N, int K, int nb)
{
    dim3 g((M + TM - 1) / TM, (N + TN - 1) / TN, nb);
    rv_gemm<<<g, 256, 0, st>>>(A, ldA, sAo, sAh, hA, Bm, ldB, sBo, sBh, hB, transB,
                               C, ldC, sCo, sCh, hC, nullptr, nullptr, 0, M, N, K, 0);
}

struct EncLayer { const float *Wo,*Wqkv,*bo,*fW1,*fW2,*fb1,*fb2,*ln1_b,*ln1_g,*ln2_b,*ln2_g; };
struct MixLayer { const float *Wkv,*Wo,*Wq,*bo,*fW1,*fW2,*fb1,*fb2,*ln2_b,*ln2_g,*lns_b,*lns_g,*lnt_b,*lnt_g; };
struct PlainTf  { EncLayer L[4]; const float *lnf_b,*lnf_g,*pos; };

extern "C" void kernel_launch(void* const* d_in, const int* in_sizes, int n_in,
                              void* d_out, int out_size, void* d_ws, size_t ws_size,
                              hipStream_t stream)
{
    (void)in_sizes; (void)n_in; (void)out_size; (void)ws_size;

    const float* ctx        = (const float*)d_in[0];
    const float* ctx_coords = (const float*)d_in[1];
    const float* ts         = (const float*)d_in[2];
    const float* ts_coords  = (const float*)d_in[3];
    const float* timec      = (const float*)d_in[4];

    int pi = 5;
    auto nx = [&]() { return (const float*)d_in[pi++]; };

    EncLayer ce[4];
    for (int l = 0; l < 4; ++l) {
        EncLayer& e = ce[l];
        e.Wo = nx(); e.Wqkv = nx(); e.bo = nx();
        e.fW1 = nx(); e.fW2 = nx(); e.fb1 = nx(); e.fb2 = nx();
        e.ln1_b = nx(); e.ln1_g = nx(); e.ln2_b = nx(); e.ln2_g = nx();
    }
    const float* dec_W = nx(); const float* dec_b = nx();
    const float* hW = nx(); const float* hb = nx(); const float* hbias = nx(); const float* hg = nx();
    MixLayer mx[4];
    for (int l = 0; l < 4; ++l) {
        MixLayer& e = mx[l];
        e.Wkv = nx(); e.Wo = nx(); e.Wq = nx(); e.bo = nx();
        e.fW1 = nx(); e.fW2 = nx(); e.fb1 = nx(); e.fb2 = nx();
        e.ln2_b = nx(); e.ln2_g = nx(); e.lns_b = nx(); e.lns_g = nx();
        e.lnt_b = nx(); e.lnt_g = nx();
    }
    const float* patch_W = nx(); const float* patch_b = nx();
    const float* qmW1 = nx(); const float* qmW2 = nx(); const float* qmW3 = nx();
    const float* qmb1 = nx(); const float* qmb2 = nx(); const float* qmb3 = nx();
    const float* qmln_b = nx(); const float* qmln_g = nx();
    const float* scales = nx();
    PlainTf temporal;
    for (int l = 0; l < 4; ++l) {
        EncLayer& e = temporal.L[l];
        e.Wo = nx(); e.Wqkv = nx(); e.bo = nx();
        e.fW1 = nx(); e.fW2 = nx(); e.fb1 = nx(); e.fb2 = nx();
        e.ln1_b = nx(); e.ln1_g = nx(); e.ln2_b = nx(); e.ln2_g = nx();
    }
    temporal.lnf_b = nx(); temporal.lnf_g = nx(); temporal.pos = nx();
    const float* ts_W = nx(); const float* ts_b = nx();
    PlainTf tsenc;
    for (int l = 0; l < 4; ++l) {
        EncLayer& e = tsenc.L[l];
        e.Wo = nx(); e.Wqkv = nx(); e.bo = nx();
        e.fW1 = nx(); e.fW2 = nx(); e.fb1 = nx(); e.fb2 = nx();
        e.ln1_b = nx(); e.ln1_g = nx(); e.ln2_b = nx(); e.ln2_g = nx();
    }
    tsenc.lnf_b = nx(); tsenc.lnf_g = nx(); tsenc.pos = nx();

    // ---- workspace arena (floats) ----
    float* W0 = (float*)d_ws;
    long off = 0;
    auto alloc = [&](long n) { float* p = W0 + off; off += n; return p; };
    float* xbuf  = alloc(49152L * 256);   // latent_ctx stream
    float* hbuf  = alloc(49152L * 256);   // LN scratch
    float* qkv   = alloc(49152L * 768);
    float* obuf  = alloc(49152L * 256);
    float* Pbuf  = alloc(96L * 256 * 256);
    float* BIG   = alloc(49152L * 512);   // patchA chunks / FF-u chunks / mixer kv
    float* TBUF  = alloc(4096L * 1024);   // GLU gated chunk
    float* sinS  = alloc(4L * 256 * 32);
    float* cosS  = alloc(4L * 256 * 32);
    float* sinTt = alloc(4L * 32);
    float* cosTt = alloc(4L * 32);
    float* tsA   = alloc(192L * 12);
    float* ts_x  = alloc(192L * 256);
    float* ts_h  = alloc(192L * 256);
    float* tqkv  = alloc(192L * 2304);
    float* Sbuf  = alloc(24L * 48 * 48);
    float* to_   = alloc(192L * 768);
    float* tu_   = alloc(192L * 2048);
    float* tt_   = alloc(192L * 1024);
    float* tq    = alloc(192L * 256);
    float* qb    = alloc(192L * 256);
    float* tgt   = alloc(192L * 256);
    float* catt  = alloc(768L * 256);
    float* ybuf  = alloc(192L * 128);
    float* yfin  = alloc(192L * 128);
    float* c1    = alloc(192L * 256);
    float* c2    = alloc(192L * 256);

    // ---- output regions (return order: outputs, quantile_mask, sattn, cattn)
    float* out_outputs = (float*)d_out;
    float* out_qmask   = out_outputs + 192;
    float* out_sattn   = out_qmask + 192;
    float* out_cattn   = out_sattn + 50331648L;

    // ---- RoPE tables ----
    rv_rope_tables<<<(4 * 256 * 32 + 255) / 256, 256, 0, stream>>>(ctx_coords, scales, sinS, cosS, 16);
    rv_rope_tables<<<1, 256, 0, stream>>>(ts_coords, scales, sinTt, cosTt, 1);

    // ---- patch embedding (chunked) ----
    for (int r0 = 0; r0 < 49152; r0 += 8192) {
        rv_build_patchA<<<(8192L * 704 + 255) / 256, 256, 0, stream>>>(ctx, timec, BIG, r0, 8192);
        mm(stream, BIG, 704, patch_W, xbuf + (long)r0 * 256, 8192, 256, 704, patch_b, nullptr, 0);
    }

    // ---- vision encoder (4 layers) ----
    for (int l = 0; l < 4; ++l) {
        const EncLayer& e = ce[l];
        rv_layernorm<<<49152, 128, 0, stream>>>(xbuf, hbuf, e.ln1_g, e.ln1_b, 256);
        mm(stream, hbuf, 256, e.Wqkv, qkv, 49152, 768, 256, nullptr, nullptr, 0);
        rv_rope_apply<<<(49152L * 128 + 255) / 256, 256, 0, stream>>>(qkv, 49152, 768, 0,   sinS, cosS, 12288, 256);
        rv_rope_apply<<<(49152L * 128 + 255) / 256, 256, 0, stream>>>(qkv, 49152, 768, 256, sinS, cosS, 12288, 256);
        for (int zb = 0; zb < 768; zb += 96) {
            float* Pout = (l == 3) ? (out_sattn + (long)zb * 65536) : Pbuf;
            const float* Abase = qkv + (long)(zb / 4) * 196608;
            // S = Q @ K^T  (batched NT)
            gemmB(stream, Abase, 768, 196608, 64, 4,
                          Abase + 256, 768, 196608, 64, 4, 1,
                          Pout, 256, 65536, 0, 1, 256, 256, 64, 96);
            rv_softmax<<<96 * 256, 256, 0, stream>>>(Pout, 256, 0.125f);
            // O = P @ V  (batched NN)
            gemmB(stream, Pout, 256, 65536, 0, 1,
                          Abase + 512, 768, 196608, 64, 4, 0,
                          obuf + (long)(zb / 4) * 65536, 256, 65536, 64, 4,
                          256, 64, 256, 96);
        }
        mm(stream, obuf, 256, e.Wo, xbuf, 49152, 256, 256, e.bo, xbuf, 0);
        rv_layernorm<<<49152, 128, 0, stream>>>(xbuf, hbuf, e.ln2_g, e.ln2_b, 256);
        for (int r0 = 0; r0 < 49152; r0 += 4096) {
            mm(stream, hbuf + (long)r0 * 256, 256, e.fW1, BIG, 4096, 2048, 256, e.fb1, nullptr, 0);
            rv_glu<<<(4096L * 1024 + 255) / 256, 256, 0, stream>>>(BIG, TBUF, 4096L * 1024, 1024);
            mm(stream, TBUF, 1024, e.fW2, xbuf + (long)r0 * 256, 4096, 256, 1024,
               e.fb2, xbuf + (long)r0 * 256, 0);
        }
    }

    // ---- ts branch: embed + plain transformer ----
    rv_build_tsA<<<(192 * 11 + 255) / 256, 256, 0, stream>>>(ts, timec, tsA);
    mm(stream, tsA, 11, ts_W, ts_x, 192, 256, 11, ts_b, nullptr, 0);
    rv_add_pos<<<(192 * 256 + 255) / 256, 256, 0, stream>>>(ts_x, tsenc.pos, 192, 256, 48);
    for (int l = 0; l < 4; ++l) {
        const EncLayer& e = tsenc.L[l];
        rv_layernorm<<<192, 128, 0, stream>>>(ts_x, ts_h, e.ln1_g, e.ln1_b, 256);
        mm(stream, ts_h, 256, e.Wqkv, tqkv, 192, 768, 256, nullptr, nullptr, 0);
        gemmB(stream, tqkv, 768, 36864, 64, 4,
                      tqkv + 256, 768, 36864, 64, 4, 1,
                      Sbuf, 48, 2304, 0, 1, 48, 48, 64, 16);
        rv_softmax<<<16 * 48, 256, 0, stream>>>(Sbuf, 48, 0.125f);
        gemmB(stream, Sbuf, 48, 2304, 0, 1,
                      tqkv + 512, 768, 36864, 64, 4, 0,
                      to_, 256, 12288, 64, 4, 48, 64, 48, 16);
        mm(stream, to_, 256, e.Wo, ts_x, 192, 256, 256, e.bo, ts_x, 0);
        rv_layernorm<<<192, 128, 0, stream>>>(ts_x, ts_h, e.ln2_g, e.ln2_b, 256);
        mm(stream, ts_h, 256, e.fW1, tu_, 192, 1024, 256, e.fb1, nullptr, 0);
        rv_gelu_inplace<<<(192L * 1024 + 255) / 256, 256, 0, stream>>>(tu_, 192L * 1024);
        mm(stream, tu_, 1024, e.fW2, ts_x, 192, 256, 1024, e.fb2, ts_x, 0);
    }
    rv_layernorm<<<192, 128, 0, stream>>>(ts_x, tgt, tsenc.lnf_g, tsenc.lnf_b, 256);

    // ---- mixer (cross-attn, 4 layers) ----
    for (int l = 0; l < 4; ++l) {
        const MixLayer& e = mx[l];
        rv_layernorm<<<49152, 128, 0, stream>>>(xbuf, hbuf, e.lns_g, e.lns_b, 256);
        mm(stream, hbuf, 256, e.Wkv, BIG, 49152, 512, 256, nullptr, nullptr, 0);
        rv_rope_apply<<<(49152L * 128 + 255) / 256, 256, 0, stream>>>(BIG, 49152, 512, 0, sinS, cosS, 12288, 256);
        rv_layernorm<<<192, 128, 0, stream>>>(tgt, tq, e.lnt_g, e.lnt_b, 256);
        mm(stream, tq, 256, e.Wq, qb, 192, 256, 256, nullptr, nullptr, 0);
        rv_rope_apply<<<(192L * 128 + 255) / 256, 256, 0, stream>>>(qb, 192, 256, 0, sinTt, cosTt, 48, 1);
        float* probs = (l == 3) ? out_cattn : catt;
        rv_cross_attn<<<768, 256, 0, stream>>>(qb, BIG, to_, probs, 4);
        mm(stream, to_, 256, e.Wo, tgt, 192, 256, 256, e.bo, tgt, 0);
        rv_layernorm<<<192, 128, 0, stream>>>(tgt, tq, e.ln2_g, e.ln2_b, 256);
        mm(stream, tq, 256, e.fW1, tu_, 192, 2048, 256, e.fb1, nullptr, 0);
        rv_glu<<<(192L * 1024 + 255) / 256, 256, 0, stream>>>(tu_, tt_, 192L * 1024, 1024);
        mm(stream, tt_, 1024, e.fW2, tgt, 192, 256, 1024, e.fb2, tgt, 0);
    }

    // ---- decode + temporal transformer ----
    mm(stream, tgt, 256, dec_W, ybuf, 192, 128, 256, dec_b, nullptr, 0);
    rv_add_pos<<<(192 * 128 + 255) / 256, 256, 0, stream>>>(ybuf, temporal.pos, 192, 128, 48);
    for (int l = 0; l < 4; ++l) {
        const EncLayer& e = temporal.L[l];
        rv_layernorm<<<192, 128, 0, stream>>>(ybuf, c1, e.ln1_g, e.ln1_b, 128);
        mm(stream, c1, 128, e.Wqkv, tqkv, 192, 2304, 128, nullptr, nullptr, 0);
        gemmB(stream, tqkv, 2304, 110592, 128, 6,
                      tqkv + 768, 2304, 110592, 128, 6, 1,
                      Sbuf, 48, 2304, 0, 1, 48, 48, 128, 24);
        rv_softmax<<<24 * 48, 256, 0, stream>>>(Sbuf, 48, 0.0883883476f);
        gemmB(stream, Sbuf, 48, 2304, 0, 1,
                      tqkv + 1536, 2304, 110592, 128, 6, 0,
                      to_, 768, 36864, 128, 6, 48, 128, 48, 24);
        mm(stream, to_, 768, e.Wo, ybuf, 192, 128, 768, e.bo, ybuf, 0);
        rv_layernorm<<<192, 128, 0, stream>>>(ybuf, c1, e.ln2_g, e.ln2_b, 128);
        mm(stream, c1, 128, e.fW1, tu_, 192, 512, 128, e.fb1, nullptr, 0);
        rv_gelu_inplace<<<(192L * 512 + 255) / 256, 256, 0, stream>>>(tu_, 192L * 512);
        mm(stream, tu_, 512, e.fW2, ybuf, 192, 128, 512, e.fb2, ybuf, 0);
    }
    rv_layernorm<<<192, 128, 0, stream>>>(ybuf, yfin, temporal.lnf_g, temporal.lnf_b, 128);

    // ---- output head ----
    rv_layernorm<<<192, 128, 0, stream>>>(yfin, c1, hg, hb, 128);
    mm(stream, c1, 128, hW, out_outputs, 192, 1, 128, hbias, nullptr, 1);

    // ---- quantile mask branch (conv1d x2 + LN + linear) ----
    rv_conv1d_relu<<<(4L * 48 * 256 + 255) / 256, 256, 0, stream>>>(yfin, qmW1, qmb1, c1, 4, 48, 128, 256);
    rv_conv1d_relu<<<(4L * 48 * 256 + 255) / 256, 256, 0, stream>>>(c1, qmW2, qmb2, c2, 4, 48, 256, 256);
    rv_layernorm<<<192, 128, 0, stream>>>(c2, c1, qmln_g, qmln_b, 256);
    mm(stream, c1, 256, qmW3, out_qmask, 192, 1, 256, qmb3, nullptr, 0);
}
